// GCN_89472758710571
// MI455X (gfx1250) — matrix operand order repaired and checked by reading, not compiled, not run on hardware
//
#include <hip/hip_runtime.h>
#include <math.h>

typedef __attribute__((ext_vector_type(16))) __bf16    v16bf;
typedef __attribute__((ext_vector_type(2)))  __bf16    v2bf;
typedef __attribute__((ext_vector_type(8)))  float     v8f;
typedef __attribute__((ext_vector_type(4)))  unsigned  uint4v;

#define F_IN    500
#define HID     128
#define NCLS    16
#define BM      128
#define BK      32
#define APAD    40   // padded LDS row stride (bf16 units) -> conflict-free b128 frag loads
#define BPAD    40

union FragBF { v16bf v; uint4v q[2]; };

// ---------------------------------------------------------------- utilities
__global__ void zero4_kernel(float4* __restrict__ p, long long n4) {
    long long i = (long long)blockIdx.x * blockDim.x + threadIdx.x;
    if (i < n4) p[i] = make_float4(0.f, 0.f, 0.f, 0.f);
}

__global__ void deg_kernel(const long long* __restrict__ dst, long long E,
                           unsigned* __restrict__ deg) {
    long long e = (long long)blockIdx.x * blockDim.x + threadIdx.x;
    if (e < E) atomicAdd(&deg[(int)dst[e]], 1u);
}

__global__ void dinv_kernel(unsigned* deg_in, float* dinv_out, int N) {
    int i = blockIdx.x * blockDim.x + threadIdx.x;
    if (i < N) {
        float d = (float)deg_in[i] + 1.0f;   // +1 = self loop; always > 0
        dinv_out[i] = rsqrtf(d);
    }
}

// ---------------------------------------------------------------- WMMA GEMM
// C[M x 128] = f(A[M x K]) @ B[K x 128]
// f = BIAS_RELU ? relu(a + in_bias[k]) : a.   Requires K even (500 / 128 here).
template<bool BIAS_RELU>
__global__ __launch_bounds__(256) void gemm_bf16_wmma(
    const float* __restrict__ A, int M, int K,
    const float* __restrict__ B,
    const float* __restrict__ in_bias,
    float* __restrict__ C)
{
    __shared__ __bf16 As [BM * APAD];   // 10240 B
    __shared__ __bf16 Bts[HID * BPAD];  // 10240 B  (B transposed: [n][k])

    const int tid  = threadIdx.x;
    const int wave = tid >> 5;
    const int lane = tid & 31;
    const int rowBase = blockIdx.x * BM;

    v8f acc[8] = {};

    const int kTiles = (K + BK - 1) / BK;
    for (int kt = 0; kt < kTiles; ++kt) {
        const int kbase = kt * BK;

        // --- stage A tile (128x32 fp32 -> bf16): branch-free float2 loads.
        // K even => a valid pair never straddles K; clamp + select, no exec masking.
        #pragma unroll
        for (int i = 0; i < 8; ++i) {
            int idx2 = i * 256 + tid;            // 2048 float2 total
            int r  = idx2 >> 4;
            int c2 = (idx2 & 15) * 2;
            int gr = rowBase + r;
            int gk = kbase + c2;
            int grc = (gr < M) ? gr : (M - 1);
            int gkc = (gk < K) ? gk : 0;
            float2 p = *(const float2*)(A + (size_t)grc * K + gkc);
            float v0 = p.x, v1 = p.y;
            if (BIAS_RELU) {
                v0 = fmaxf(v0 + in_bias[gkc],     0.f);
                v1 = fmaxf(v1 + in_bias[gkc + 1], 0.f);
            }
            bool ok = (gr < M) && (gk < K);
            v0 = ok ? v0 : 0.f;
            v1 = ok ? v1 : 0.f;
            v2bf st; st.x = (__bf16)v0; st.y = (__bf16)v1;
            *(v2bf*)&As[r * APAD + c2] = st;     // 4B-aligned packed store
        }

        // --- stage B tile transposed (Bts[n][k] = B[kbase+k][n]): float4 loads
        // (B rows are 128 floats -> always 16B-aligned).
        #pragma unroll
        for (int i = 0; i < 4; ++i) {
            int idx4 = i * 256 + tid;            // 1024 float4 total
            int k  = idx4 >> 5;
            int n4 = (idx4 & 31) * 4;
            int gk = kbase + k;
            int gkc = (gk < K) ? gk : 0;
            float4 p = *(const float4*)(B + (size_t)gkc * HID + n4);
            bool ok = (gk < K);
            float vs0 = ok ? p.x : 0.f, vs1 = ok ? p.y : 0.f;
            float vs2 = ok ? p.z : 0.f, vs3 = ok ? p.w : 0.f;
            Bts[(n4 + 0) * BPAD + k] = (__bf16)vs0;
            Bts[(n4 + 1) * BPAD + k] = (__bf16)vs1;
            Bts[(n4 + 2) * BPAD + k] = (__bf16)vs2;
            Bts[(n4 + 3) * BPAD + k] = (__bf16)vs3;
        }
        __syncthreads();

        // --- A fragment for this wave (rows wave*16..+15), ISA 16-bit A layout
        const int arow = wave * 16 + (lane & 15);
        const int ak0  = (lane < 16) ? 0 : 8;
        FragBF fa;
        fa.q[0] = *(const uint4v*)&As[arow * APAD + ak0];
        fa.q[1] = *(const uint4v*)&As[arow * APAD + ak0 + 16];

        const int bk0 = (lane < 16) ? 0 : 8;     // ISA 16-bit B layout
        #pragma unroll
        for (int nt = 0; nt < 8; ++nt) {
            const int bn = nt * 16 + (lane & 15);
            FragBF fb;
            fb.q[0] = *(const uint4v*)&Bts[bn * BPAD + bk0];
            fb.q[1] = *(const uint4v*)&Bts[bn * BPAD + bk0 + 16];
            acc[nt] = __builtin_amdgcn_wmma_f32_16x16x32_bf16(
                false, fa.v, false, fb.v, (short)0, acc[nt], false, false);
        }
        __syncthreads();
    }

    // --- epilogue: C layout (VGPR r: lanes0-15 -> M=r, lanes16-31 -> M=r+8).
    // Uniform fast path: 781/782 blocks are full -> unguarded coalesced stores.
    const int rbase = rowBase + wave * 16 + ((lane < 16) ? 0 : 8);
    const int col0  = lane & 15;
    if (rowBase + BM <= M) {
        #pragma unroll
        for (int nt = 0; nt < 8; ++nt) {
            float* cp = C + (size_t)rbase * HID + nt * 16 + col0;
            #pragma unroll
            for (int r = 0; r < 8; ++r)
                cp[(size_t)r * HID] = acc[nt][r];
        }
    } else {
        #pragma unroll
        for (int nt = 0; nt < 8; ++nt) {
            const int col = nt * 16 + col0;
            #pragma unroll
            for (int r = 0; r < 8; ++r) {
                int row = rbase + r;
                if (row < M) C[(size_t)row * HID + col] = acc[nt][r];
            }
        }
    }
}

// ---------------------------------------------------------------- scatter-add
// One wave per edge; lane handles a float4 slice. Out lives in L2 (51 MB << 192 MB).
__global__ __launch_bounds__(256) void agg_kernel(
    const long long* __restrict__ src, const long long* __restrict__ dst,
    long long E, int N, const float* __restrict__ dinv,
    const float* __restrict__ H, float* __restrict__ Out)
{
    long long gid = (long long)blockIdx.x * blockDim.x + threadIdx.x;
    long long e = gid >> 5;
    int lane = (int)(gid & 31);
    long long total = E + N;
    if (e >= total) return;
    int s, d;
    if (e < E) { s = (int)src[e]; d = (int)dst[e]; }
    else       { s = d = (int)(e - E); }               // self loop
    float norm = dinv[s] * dinv[d];
    float4 v = ((const float4*)(H + (size_t)s * HID))[lane];
    float* o = Out + (size_t)d * HID + lane * 4;
    atomicAdd(o + 0, v.x * norm);
    atomicAdd(o + 1, v.y * norm);
    atomicAdd(o + 2, v.z * norm);
    atomicAdd(o + 3, v.w * norm);
}

// ---------------------------------------------------------------- FC + softmax
__global__ __launch_bounds__(256) void fc_softmax_kernel(
    const float* __restrict__ Agg2, const float* __restrict__ b2,
    const float* __restrict__ Wfc, const float* __restrict__ bfc,
    float* __restrict__ out, int M)
{
    __shared__ float Wsh[HID * NCLS];
    __shared__ float b2sh[HID];
    __shared__ float bfcsh[NCLS];
    int tid = threadIdx.x;
    for (int i = tid; i < HID * NCLS; i += 256) Wsh[i] = Wfc[i];
    if (tid < HID)  b2sh[tid]  = b2[tid];
    if (tid < NCLS) bfcsh[tid] = bfc[tid];
    __syncthreads();

    int node = blockIdx.x * 256 + tid;
    if (node >= M) return;

    float lg[NCLS];
    #pragma unroll
    for (int c = 0; c < NCLS; ++c) lg[c] = bfcsh[c];

    const float* a = Agg2 + (size_t)node * HID;
    #pragma unroll 4
    for (int k = 0; k < HID; ++k) {
        float v = fmaxf(a[k] + b2sh[k], 0.f);
        #pragma unroll
        for (int c = 0; c < NCLS; ++c) lg[c] = fmaf(v, Wsh[k * NCLS + c], lg[c]);
    }
    float m = lg[0];
    #pragma unroll
    for (int c = 1; c < NCLS; ++c) m = fmaxf(m, lg[c]);
    float sum = 0.f;
    #pragma unroll
    for (int c = 0; c < NCLS; ++c) { lg[c] = __expf(lg[c] - m); sum += lg[c]; }
    float inv = 1.f / sum;
    #pragma unroll
    for (int c = 0; c < NCLS; ++c) out[(size_t)node * NCLS + c] = lg[c] * inv;
}

// ---------------------------------------------------------------- launch
extern "C" void kernel_launch(void* const* d_in, const int* in_sizes, int n_in,
                              void* d_out, int out_size, void* d_ws, size_t ws_size,
                              hipStream_t stream) {
    const float*     x   = (const float*)d_in[0];
    const long long* ei  = (const long long*)d_in[1];   // int64 edge_index (2,E)
    const float*     W1  = (const float*)d_in[2];
    const float*     b1  = (const float*)d_in[3];
    const float*     W2  = (const float*)d_in[4];
    const float*     b2  = (const float*)d_in[5];
    const float*     Wfc = (const float*)d_in[6];
    const float*     bfc = (const float*)d_in[7];
    float*           out = (float*)d_out;

    const int       M = in_sizes[0] / F_IN;        // 100000
    const long long E = (long long)in_sizes[1] / 2;
    const long long* srcp = ei;
    const long long* dstp = ei + E;

    // workspace carve: dinv/deg | bufA | bufB
    char* ws = (char*)d_ws;
    float* dinv = (float*)ws;
    size_t ofsA = ((size_t)M * 4 + 255) & ~(size_t)255;
    float* bufA = (float*)(ws + ofsA);
    float* bufB = bufA + (size_t)M * HID;

    const int T = 256;
    const long long degN4 = ((long long)M * 4 + 15) / 16;     // deg region in float4s
    const long long hN4   = (long long)M * HID / 4;           // 128-wide buffer in float4s
    const long long aggT  = (E + M) * 32;                     // wave per edge
    const int gridGemm = (M + BM - 1) / BM;

    // degree (int atomics on zeroed region), then dinv = rsqrt(deg+1)
    zero4_kernel<<<(int)((degN4 + T - 1) / T), T, 0, stream>>>((float4*)dinv, degN4);
    deg_kernel  <<<(int)((E + T - 1) / T),    T, 0, stream>>>(dstp, E, (unsigned*)dinv);
    dinv_kernel <<<(M + T - 1) / T,           T, 0, stream>>>((unsigned*)dinv, dinv, M);

    // layer 1: H = X @ W1 ; Agg1 = scatter(H * norm)
    gemm_bf16_wmma<false><<<gridGemm, T, 0, stream>>>(x, M, F_IN, W1, nullptr, bufA);
    zero4_kernel<<<(int)((hN4 + T - 1) / T), T, 0, stream>>>((float4*)bufB, hN4);
    agg_kernel  <<<(int)((aggT + T - 1) / T), T, 0, stream>>>(srcp, dstp, E, M, dinv, bufA, bufB);

    // layer 2: H = relu(Agg1 + b1) @ W2 (bias+relu fused into A staging) ; Agg2
    gemm_bf16_wmma<true><<<gridGemm, T, 0, stream>>>(bufB, M, HID, W2, b1, bufA);
    zero4_kernel<<<(int)((hN4 + T - 1) / T), T, 0, stream>>>((float4*)bufB, hN4);
    agg_kernel  <<<(int)((aggT + T - 1) / T), T, 0, stream>>>(srcp, dstp, E, M, dinv, bufA, bufB);

    // head: softmax(relu(Agg2 + b2) @ Wfc + bfc)
    fc_softmax_kernel<<<(M + T - 1) / T, T, 0, stream>>>(bufB, b2, Wfc, bfc, out, M);
}
